// SheetBase_69406671503665
// MI455X (gfx1250) — matrix operand order repaired and checked by reading, not compiled, not run on hardware
//
#include <hip/hip_runtime.h>

typedef _Float16 h8   __attribute__((ext_vector_type(8)));
typedef _Float16 h16  __attribute__((ext_vector_type(16)));
typedef float    f8   __attribute__((ext_vector_type(8)));

#define MROWS 64
#define S0 72          // h0 row stride in halves (64 used + 8 pad)
#define S1 520         // h1/h2 row stride in halves (512 used + 8 pad)
#define TSZ (1u << 19)
#define CLAMPV 25.133f
#define INV_2PI 0.15915494309189535f

__constant__ int RES_TBL[16] = {32,44,61,84,116,161,222,307,425,588,812,1123,1551,2144,2963,4095};

__device__ __forceinline__ h16 cat8(h8 lo, h8 hi) {
  return __builtin_shufflevector(lo, hi, 0,1,2,3,4,5,6,7,8,9,10,11,12,13,14,15);
}

// sin(clip(v, -CLAMPV, CLAMPV)) via v_med3_f32 + v_sin_f32.
// |v| <= 25.133 -> |v/(2pi)| <= 4.001, well inside V_SIN_F32's valid range.
__device__ __forceinline__ float act_sin(float v) {
  v = __builtin_amdgcn_fmed3f(v, -CLAMPV, CLAMPV);
  return __builtin_amdgcn_sinf(v * INV_2PI);
}

// Compute a 4(M) x NT(N) grid of 16x16 WMMA tiles: act(64 x KD, f16 in LDS) @ W^T + bias.
// W is row-major f16 (out x KD) in global memory (L2 resident).
// Accumulators are initialized with the bias (uniform across each tile column).
template<int NT, int KD, int INSTRIDE>
__device__ __forceinline__ void gemm4xN(const _Float16* actIn,
                                        const _Float16* __restrict__ W,
                                        const float* __restrict__ bias,
                                        int nbase, int l16, int lhalf,
                                        f8 (&acc)[4][NT]) {
#pragma unroll
  for (int n = 0; n < NT; ++n) {
    float bv = bias[nbase + n * 16 + l16];   // C/D layout: column = lane&15
#pragma unroll
    for (int m = 0; m < 4; ++m)
#pragma unroll
      for (int i = 0; i < 8; ++i) acc[m][n][i] = bv;
  }

#pragma unroll 2
  for (int ks = 0; ks < KD / 32; ++ks) {
    // B fragments: lane holds column nbase+n*16+(lane&15); halves are
    // K = ks*32 + lhalf*16 + 0..15 (contiguous).
    h16 bf[NT];
#pragma unroll
    for (int n = 0; n < NT; ++n) {
      const _Float16* q = W + (size_t)(nbase + n * 16 + l16) * KD + ks * 32 + lhalf * 16;
      bf[n] = cat8(*(const h8*)q, *(const h8*)(q + 8));
    }
#pragma unroll
    for (int m = 0; m < 4; ++m) {
      // A fragment: lane holds row m*16+(lane&15); halves are
      // K = ks*32 + lhalf*16 + 0..15 (contiguous).
      const _Float16* p = actIn + (m * 16 + l16) * INSTRIDE + ks * 32 + lhalf * 16;
      h16 af = cat8(*(const h8*)p, *(const h8*)(p + 8));
#pragma unroll
      for (int n = 0; n < NT; ++n)
        acc[m][n] = __builtin_amdgcn_wmma_f32_16x16x32_f16(
            false, af, false, bf[n], (short)0, acc[m][n], false, false);
    }
  }
}

// clip + sin, store f16 activations into LDS for the next layer.
template<int NT>
__device__ __forceinline__ void epilogue_lds(f8 (&acc)[4][NT],
                                             _Float16* actOut, int strideOut,
                                             int nbase, int l16, int lhalf) {
#pragma unroll
  for (int n = 0; n < NT; ++n) {
    int col = nbase + n * 16 + l16;
#pragma unroll
    for (int m = 0; m < 4; ++m)
#pragma unroll
      for (int i = 0; i < 8; ++i) {
        int row = m * 16 + 8 * lhalf + i;    // C/D layout: VGPR i -> M = i + 8*(lane>=16)
        actOut[row * strideOut + col] = (_Float16)act_sin(acc[m][n][i]);
      }
  }
}

__global__ void prep_weights(const float* __restrict__ w1, const float* __restrict__ w2,
                             const float* __restrict__ w3,
                             _Float16* __restrict__ w1h, _Float16* __restrict__ w2h,
                             _Float16* __restrict__ w3h) {
  int i = blockIdx.x * 256 + threadIdx.x;
  if (i < 512 * 64)  w1h[i] = (_Float16)w1[i];
  if (i < 512 * 512) w2h[i] = (_Float16)w2[i];
  if (i < 384 * 512) w3h[i] = (_Float16)w3[i];
}

__global__ __launch_bounds__(256)
void mlp_kernel(const float* __restrict__ y, const float* __restrict__ x,
                const float* __restrict__ cam, const float* __restrict__ table,
                const float* __restrict__ b1, const float* __restrict__ b2,
                const float* __restrict__ b3,
                const _Float16* __restrict__ w1h, const _Float16* __restrict__ w2h,
                const _Float16* __restrict__ w3h, float* __restrict__ out) {
  __shared__ __align__(16) _Float16 smA[MROWS * S1];  // h0 (stride S0), later h2 (stride S1)
  __shared__ __align__(16) _Float16 smB[MROWS * S1];  // h1 (stride S1)

  const int tid  = threadIdx.x;
  const int row0 = blockIdx.x * MROWS;

  // ---- Stage 0: build 64-dim f16 input features in smA (stride S0) ----
  // cam_feat -> cols [32,64)
  for (int idx = tid; idx < MROWS * 32; idx += 256) {
    int r = idx >> 5, c = idx & 31;
    smA[r * S0 + 32 + c] = (_Float16)cam[(size_t)(row0 + r) * 32 + c];
  }
  // hashgrid -> cols [0,32): (row, level) per iteration
  for (int idx = tid; idx < MROWS * 16; idx += 256) {
    int r = idx >> 4, l = idx & 15;
    float xv = x[row0 + r], yv = y[row0 + r];
    float ux = fminf(fmaxf(xv, 0.0f), 511.0f) * (1.0f / 511.0f);
    float uy = fminf(fmaxf(yv, 0.0f), 511.0f) * (1.0f / 511.0f);
    int res = RES_TBL[l];
    float fr = (float)(res - 1);
    float px = ux * fr, py = uy * fr;
    float p0x = floorf(px), p0y = floorf(py);
    float fx = px - p0x, fy = py - p0y;
    int x0 = (int)p0x, y0 = (int)p0y;
    int rm1 = res - 1;
    int x1 = (x0 + 1 < rm1) ? x0 + 1 : rm1;
    int y1 = (y0 + 1 < rm1) ? y0 + 1 : rm1;
    const unsigned P = 2654435761u, M = TSZ - 1u;
    unsigned hx0 = (unsigned)x0, hx1 = (unsigned)x1;
    unsigned hy0 = (unsigned)y0 * P, hy1 = (unsigned)y1 * P;
    const float2* tab = (const float2*)table + (size_t)l * TSZ;
    float2 c00 = tab[(hx0 ^ hy0) & M];
    float2 c10 = tab[(hx1 ^ hy0) & M];
    float2 c01 = tab[(hx0 ^ hy1) & M];
    float2 c11 = tab[(hx1 ^ hy1) & M];
    float w00 = (1.0f - fx) * (1.0f - fy), w10 = fx * (1.0f - fy);
    float w01 = (1.0f - fx) * fy,          w11 = fx * fy;
    float f0 = c00.x * w00 + c10.x * w10 + c01.x * w01 + c11.x * w11;
    float f1 = c00.y * w00 + c10.y * w10 + c01.y * w01 + c11.y * w11;
    smA[r * S0 + l * 2]     = (_Float16)f0;
    smA[r * S0 + l * 2 + 1] = (_Float16)f1;
  }
  __syncthreads();

  const int lane  = tid & 31;
  const int wid   = tid >> 5;     // 8 waves
  const int l16   = lane & 15;
  const int lhalf = lane >> 4;

  // ---- Layer 1: 64 -> 512, smA(h0) -> smB(h1) ----
  {
    f8 acc[4][4];
    gemm4xN<4, 64, S0>(smA, w1h, b1, wid * 64, l16, lhalf, acc);
    epilogue_lds<4>(acc, smB, S1, wid * 64, l16, lhalf);
  }
  __syncthreads();

  // ---- Layer 2: 512 -> 512, smB(h1) -> smA(h2) ----
  {
    f8 acc[4][4];
    gemm4xN<4, 512, S1>(smB, w2h, b2, wid * 64, l16, lhalf, acc);
    epilogue_lds<4>(acc, smA, S1, wid * 64, l16, lhalf);
  }
  __syncthreads();

  // ---- Layer 3: 512 -> 384, smA(h2) -> global out (f32) ----
  {
    f8 acc[4][3];
    gemm4xN<3, 512, S1>(smA, w3h, b3, wid * 48, l16, lhalf, acc);
    float* op = out + (size_t)row0 * 384;
#pragma unroll
    for (int n = 0; n < 3; ++n) {
      int col = wid * 48 + n * 16 + l16;
#pragma unroll
      for (int m = 0; m < 4; ++m)
#pragma unroll
        for (int i = 0; i < 8; ++i) {
          int row = m * 16 + 8 * lhalf + i;
          op[(size_t)row * 384 + col] = act_sin(acc[m][n][i]);
        }
    }
  }
}

extern "C" void kernel_launch(void* const* d_in, const int* in_sizes, int n_in,
                              void* d_out, int out_size, void* d_ws, size_t ws_size,
                              hipStream_t stream) {
  const float* y     = (const float*)d_in[0];
  const float* x     = (const float*)d_in[1];
  const float* cam   = (const float*)d_in[2];
  const float* table = (const float*)d_in[3];
  const float* w1    = (const float*)d_in[4];
  const float* b1    = (const float*)d_in[5];
  const float* w2    = (const float*)d_in[6];
  const float* b2    = (const float*)d_in[7];
  const float* w3    = (const float*)d_in[8];
  const float* b3    = (const float*)d_in[9];

  _Float16* w1h = (_Float16*)d_ws;                                   // 512*64  = 64 KB
  _Float16* w2h = (_Float16*)((char*)d_ws + 65536);                  // 512*512 = 512 KB
  _Float16* w3h = (_Float16*)((char*)d_ws + 65536 + 524288);         // 384*512 = 384 KB

  prep_weights<<<1024, 256, 0, stream>>>(w1, w2, w3, w1h, w2h, w3h);

  const int nrows = in_sizes[0];             // 2^19
  mlp_kernel<<<nrows / MROWS, 256, 0, stream>>>(y, x, cam, table, b1, b2, b3,
                                                w1h, w2h, w3h, (float*)d_out);
}
